// GraphEncoder_979252543764
// MI455X (gfx1250) — compile-verified, matrix operand order checked by reading, not run on hardware
//
#include <hip/hip_runtime.h>
#include <hip/hip_bf16.h>
#include <math.h>

// Problem constants (match reference)
#define GB   512          // graphs
#define GN   256          // nodes per graph
#define GE   1024         // edges per graph
#define FIN  32
#define NH   128
#define BN   (GB*GN)      // 131072
#define BE   (GB*GE)      // 524288

typedef __attribute__((ext_vector_type(16))) __bf16 v16bf;
typedef __attribute__((ext_vector_type(8)))  __bf16 v8bf;
typedef __attribute__((ext_vector_type(8)))  float  v8f;

// ---------------------------------------------------------------------------
// One-time convert: f32 -> bf16 (stage-0 GEMM A operand)
// ---------------------------------------------------------------------------
__global__ void f32_to_bf16(const float* __restrict__ in, __bf16* __restrict__ out, int n)
{
    int i = blockIdx.x * 256 + threadIdx.x;
    if (i < n) out[i] = (__bf16)in[i];
}

// ---------------------------------------------------------------------------
// GEMM: out[M x NH] = A[M x K] @ W[K x NH]  (A bf16, W f32->bf16, f32 accum)
// Block = 128 threads = 4 waves; each wave owns a 16-row M tile and all 8
// column tiles. W is pre-packed into LDS in the exact WMMA B-fragment
// layout: [ktile][ntile][lane][16 bf16] -> each fragment is two ds_load_b128.
// A fragment is two aligned global 16B bf16 loads (no conversion in loop).
// ---------------------------------------------------------------------------
template<int K>
__global__ __launch_bounds__(128)
void gemm_wmma_bf16(const __bf16* __restrict__ A, const float* __restrict__ W,
                    float* __restrict__ out)
{
    __shared__ __attribute__((aligned(32))) __bf16 Wp[K * NH];
    const int t = threadIdx.x;
    // Pre-pack W into B-fragment order.
    // Packed flat index p = (((ktile*8 + ntile)*32 + lane)*16 + elem)
    //   n  = ntile*16 + (lane & 15)
    //   kb = (lane < 16) ? 0 : 8
    //   k  = ktile*32 + (elem < 8 ? kb + elem : kb + 8 + elem)
    for (int p = t; p < K * NH; p += 128) {
        int elem = p & 15;
        int lane = (p >> 4) & 31;
        int nt   = (p >> 9) & 7;
        int kkt  = p >> 12;
        int n    = nt * 16 + (lane & 15);
        int kb   = (lane < 16) ? 0 : 8;
        int kr   = (elem < 8) ? (kb + elem) : (kb + 8 + elem);
        Wp[p] = (__bf16)W[(kkt * 32 + kr) * NH + n];
    }
    __syncthreads();

    const int wave  = t >> 5;
    const int lane  = t & 31;
    const int m0    = (blockIdx.x * 4 + wave) * 16;
    const int mrow  = m0 + (lane & 15);
    const int khalf = (lane < 16) ? 0 : 8;
    const int ncol  = lane & 15;

    v8f acc[8];
#pragma unroll
    for (int j = 0; j < 8; ++j) acc[j] = {};

#pragma unroll
    for (int kkt = 0; kkt < K / 32; ++kkt) {
        // A fragment: elems 0..7 -> K=kb..kb+7, elems 8..15 -> K=kb+16..kb+23
        const __bf16* arow = A + (size_t)mrow * K + kkt * 32 + khalf;
        const v8bf alo = *(const v8bf*)(arow);
        const v8bf ahi = *(const v8bf*)(arow + 16);
        v16bf a;
#pragma unroll
        for (int i = 0; i < 8; ++i) { a[i] = alo[i]; a[8 + i] = ahi[i]; }

#pragma unroll
        for (int nt = 0; nt < 8; ++nt) {
            const v16bf bfr = *(const v16bf*)&Wp[((kkt * 8 + nt) * 32 + lane) * 16];
            acc[nt] = __builtin_amdgcn_wmma_f32_16x16x32_bf16(
                false, a, false, bfr, (short)0, acc[nt], false, false);
        }
    }

    // C layout: lanes 0-15: N=lane, M=r; lanes 16-31: N=lane-16, M=8+r
    const int msel = (lane < 16) ? 0 : 8;
#pragma unroll
    for (int nt = 0; nt < 8; ++nt) {
#pragma unroll
        for (int r = 0; r < 8; ++r) {
            out[(size_t)(m0 + msel + r) * NH + nt * 16 + ncol] = acc[nt][r];
        }
    }
}

// ---------------------------------------------------------------------------
// Degree: deg[i] = valid[i] + sum over edges into i of valid[gs]*valid[gd]
// ---------------------------------------------------------------------------
__global__ void fill_ones(float* v) {
    int i = blockIdx.x * 256 + threadIdx.x;
    if (i < BN) v[i] = 1.0f;
}
__global__ void deg_init(const float* __restrict__ valid, float* __restrict__ deg) {
    int i = blockIdx.x * 256 + threadIdx.x;
    if (i < BN) deg[i] = valid[i];
}
__global__ void deg_edges(const int* __restrict__ src, const int* __restrict__ dst,
                          const float* __restrict__ valid, float* __restrict__ deg) {
    int e = blockIdx.x * 256 + threadIdx.x;
    if (e >= BE) return;
    int b = e >> 10;                      // GE = 1024
    int gs = b * GN + src[e];
    int gd = b * GN + dst[e];
    float ev = valid[gs] * valid[gd];
    if (ev != 0.0f) atomicAdd(&deg[gd], ev);
}

// ---------------------------------------------------------------------------
// Edge message pass (vector, NH feats): one wave per edge, float4 per lane.
// coef = dis[gs]*dis[gd] with dis = deg>0 ? rsqrt(deg) : 0 (deg==0 iff invalid)
// ---------------------------------------------------------------------------
__global__ __launch_bounds__(256)
void edge_pass_vec(const float* __restrict__ hw, const int* __restrict__ src,
                   const int* __restrict__ dst, const float* __restrict__ deg,
                   float* __restrict__ agg)
{
    int gid  = blockIdx.x * 256 + threadIdx.x;
    int e    = gid >> 5;
    int lane = gid & 31;
    if (e >= BE) return;
    int b  = e >> 10;
    int gs = b * GN + src[e];
    int gd = b * GN + dst[e];
    float ds = deg[gs], dd = deg[gd];
    if (ds <= 0.0f || dd <= 0.0f) return;
    float c = rsqrtf(ds * dd);
    const float4 v = *(const float4*)(hw + (size_t)gs * NH + lane * 4);
    float* ap = agg + (size_t)gd * NH + lane * 4;
    atomicAdd(ap + 0, v.x * c);
    atomicAdd(ap + 1, v.y * c);
    atomicAdd(ap + 2, v.z * c);
    atomicAdd(ap + 3, v.w * c);
}

// ---------------------------------------------------------------------------
// Conv combine: h = relu( (agg + hw/deg + b) * valid )   (in place into agg)
// ---------------------------------------------------------------------------
__global__ void combine_relu(const float* __restrict__ hw, float* __restrict__ aggh,
                             const float* __restrict__ deg, const float* __restrict__ valid,
                             const float* __restrict__ bias)
{
    int idx = blockIdx.x * 256 + threadIdx.x;
    if (idx >= BN * NH) return;
    int node = idx >> 7;       // NH = 128
    int f    = idx & 127;
    float d  = deg[node];
    float d2 = d > 0.0f ? 1.0f / d : 0.0f;
    float o  = (aggh[idx] + hw[idx] * d2 + bias[f]) * valid[node];
    aggh[idx] = fmaxf(o, 0.0f);
}

// ---------------------------------------------------------------------------
// Score conv: hw_s[i] = dot(h[i], Ws)  (one wave per node, shuffle reduce)
// ---------------------------------------------------------------------------
__global__ __launch_bounds__(256)
void score_dot(const float* __restrict__ h, const float* __restrict__ Ws,
               float* __restrict__ hws)
{
    int gid  = blockIdx.x * 256 + threadIdx.x;
    int node = gid >> 5;
    int lane = gid & 31;
    if (node >= BN) return;
    const float4 hv = *(const float4*)(h + (size_t)node * NH + lane * 4);
    const float4 wv = *(const float4*)(Ws + lane * 4);
    float s = hv.x * wv.x + hv.y * wv.y + hv.z * wv.z + hv.w * wv.w;
#pragma unroll
    for (int off = 16; off; off >>= 1) s += __shfl_xor(s, off, 32);
    if (lane == 0) hws[node] = s;
}

__global__ void edge_pass_scalar(const float* __restrict__ hws, const int* __restrict__ src,
                                 const int* __restrict__ dst, const float* __restrict__ deg,
                                 float* __restrict__ aggs)
{
    int e = blockIdx.x * 256 + threadIdx.x;
    if (e >= BE) return;
    int b  = e >> 10;
    int gs = b * GN + src[e];
    int gd = b * GN + dst[e];
    float ds = deg[gs], dd = deg[gd];
    if (ds <= 0.0f || dd <= 0.0f) return;
    atomicAdd(&aggs[gd], hws[gs] * rsqrtf(ds * dd));
}

__global__ void score_combine(const float* __restrict__ hws, const float* __restrict__ aggs,
                              const float* __restrict__ deg, const float* __restrict__ valid,
                              const float* __restrict__ bs, float* __restrict__ score)
{
    int i = blockIdx.x * 256 + threadIdx.x;
    if (i >= BN) return;
    float d  = deg[i];
    float d2 = d > 0.0f ? 1.0f / d : 0.0f;
    score[i] = (aggs[i] + hws[i] * d2 + bs[0]) * valid[i];
}

// ---------------------------------------------------------------------------
// Top-k per graph via O(N^2) stable rank (matches lax.top_k tie-break: lower
// index wins). One 256-thread block per graph, scores in LDS.
// ---------------------------------------------------------------------------
__global__ __launch_bounds__(256)
void topk_select(const float* __restrict__ score, const float* __restrict__ valid,
                 float* __restrict__ newvalid, int k)
{
    __shared__ float s[GN];
    int g = blockIdx.x, t = threadIdx.x;
    int idx = g * GN + t;
    float va = valid[idx];
    float ms = va > 0.0f ? score[idx] : -__builtin_inff();
    s[t] = ms;
    __syncthreads();
    int rank = 0;
    for (int j = 0; j < GN; ++j) {
        float v = s[j];
        rank += (v > ms) || (v == ms && j < t);
    }
    newvalid[idx] = (rank < k && va > 0.0f) ? 1.0f : 0.0f;
}

// ---------------------------------------------------------------------------
// Pool + readout: xn = h * tanh(score) * newvalid written directly in bf16
// (only consumer is next stage's WMMA A operand); gmax/gmean over kept nodes
// accumulated into out. One block per graph, 128 threads (one per feature).
// ---------------------------------------------------------------------------
__global__ __launch_bounds__(128)
void pool_readout(const float* __restrict__ h, const float* __restrict__ score,
                  const float* __restrict__ newvalid, __bf16* __restrict__ xn,
                  float* __restrict__ out, int k, int first)
{
    int g = blockIdx.x, f = threadIdx.x;
    float gmax = -__builtin_inff();
    float gsum = 0.0f;
    for (int n = 0; n < GN; ++n) {
        int idx = g * GN + n;
        float ts = tanhf(score[idx]);
        float v  = h[(size_t)idx * NH + f] * ts;
        float nv = newvalid[idx];
        xn[(size_t)idx * NH + f] = (__bf16)(nv > 0.0f ? v : 0.0f);
        if (nv > 0.0f) { gmax = fmaxf(gmax, v); gsum += v; }
    }
    float gmean = gsum / (float)k;
    int o0 = g * (2 * NH) + f;
    if (first) { out[o0] = gmax; out[o0 + NH] = gmean; }
    else       { out[o0] += gmax; out[o0 + NH] += gmean; }
}

// ---------------------------------------------------------------------------
extern "C" void kernel_launch(void* const* d_in, const int* in_sizes, int n_in,
                              void* d_out, int out_size, void* d_ws, size_t ws_size,
                              hipStream_t stream)
{
    const float* x   = (const float*)d_in[0];
    const int*   src = (const int*)  d_in[1];
    const int*   dst = (const int*)  d_in[2];
    const float* Wc[3] = {(const float*)d_in[3],  (const float*)d_in[7],  (const float*)d_in[11]};
    const float* bc[3] = {(const float*)d_in[4],  (const float*)d_in[8],  (const float*)d_in[12]};
    const float* Ws[3] = {(const float*)d_in[5],  (const float*)d_in[9],  (const float*)d_in[13]};
    const float* bs[3] = {(const float*)d_in[6],  (const float*)d_in[10], (const float*)d_in[14]};
    float* out = (float*)d_out;

    // Workspace carve-up:
    //   2 x 64MB f32 (hw, agg->h in place), 2 x 32MB bf16 (xn ping/pong),
    //   8MB bf16 stage-0 A, per-node scalars.
    const size_t BIGF = (size_t)BN * NH * sizeof(float);
    const size_t BIGH = (size_t)BN * NH * sizeof(__bf16);
    const size_t XBF  = (size_t)BN * FIN * sizeof(__bf16);
    const size_t SML  = (size_t)BN * sizeof(float);
    char* p = (char*)d_ws;
    float*  hwbuf = (float*)p;  p += BIGF;
    float*  aghbf = (float*)p;  p += BIGF;
    __bf16* bfA   = (__bf16*)p; p += BIGH;
    __bf16* bfB   = (__bf16*)p; p += BIGH;
    __bf16* xbf0  = (__bf16*)p; p += XBF;
    float* deg  = (float*)p; p += SML;
    float* vA   = (float*)p; p += SML;
    float* vB   = (float*)p; p += SML;
    float* hws  = (float*)p; p += SML;
    float* aggs = (float*)p; p += SML;
    float* scr  = (float*)p; p += SML;

    const __bf16* xin[3] = { xbf0, bfA, bfB };   // GEMM A per stage (bf16)
    __bf16*       xnb[3] = { bfA,  bfB, bfA };   // pool output per stage
    const int     ks[3]  = { 128, 64, 32 };      // K1, K2, K3

    f32_to_bf16<<<(BN * FIN) / 256, 256, 0, stream>>>(x, xbf0, BN * FIN);
    fill_ones  <<<BN / 256, 256, 0, stream>>>(vA);
    float* vcur = vA;
    float* vnxt = vB;

    for (int s = 0; s < 3; ++s) {
        // degrees (shared by both gcn calls this stage)
        deg_init <<<BN / 256, 256, 0, stream>>>(vcur, deg);
        deg_edges<<<BE / 256, 256, 0, stream>>>(src, dst, vcur, deg);

        // conv GCN
        hipMemsetAsync(aghbf, 0, BIGF, stream);
        if (s == 0)
            gemm_wmma_bf16<FIN><<<BN / 64, 128, 0, stream>>>(xin[s], Wc[s], hwbuf);
        else
            gemm_wmma_bf16<NH> <<<BN / 64, 128, 0, stream>>>(xin[s], Wc[s], hwbuf);
        edge_pass_vec<<<(BE * 32) / 256, 256, 0, stream>>>(hwbuf, src, dst, deg, aghbf);
        combine_relu <<<(BN * NH) / 256, 256, 0, stream>>>(hwbuf, aghbf, deg, vcur, bc[s]);

        // score GCN (Fo = 1)
        score_dot       <<<(BN * 32) / 256, 256, 0, stream>>>(aghbf, Ws[s], hws);
        hipMemsetAsync(aggs, 0, SML, stream);
        edge_pass_scalar<<<BE / 256, 256, 0, stream>>>(hws, src, dst, deg, aggs);
        score_combine   <<<BN / 256, 256, 0, stream>>>(hws, aggs, deg, vcur, bs[s], scr);

        // top-k, pool, readout
        topk_select <<<GB, 256, 0, stream>>>(scr, vcur, vnxt, ks[s]);
        pool_readout<<<GB, 128, 0, stream>>>(aghbf, scr, vnxt, xnb[s], out, ks[s], s == 0);

        float* t = vcur; vcur = vnxt; vnxt = t;
    }
}
